// FFhGRUwithGabor_28174985461922
// MI455X (gfx1250) — matrix operand order repaired and checked by reading, not compiled, not run on hardware
//
#include <hip/hip_runtime.h>
#include <math.h>

typedef __attribute__((ext_vector_type(16))) __bf16 v16bf;
typedef __attribute__((ext_vector_type(8)))  float  v8f;
typedef __attribute__((ext_vector_type(4)))  unsigned int v4u;
typedef __attribute__((ext_vector_type(8)))  int v8i;
typedef __attribute__((ext_vector_type(4)))  int v4i;

#define CCH 25
#define Wd  128
#define HW  16384        // 128*128
#define LDSW 78          // 64 + 2*7 halo
#define CTILE 1170       // 15*78 contiguous per-channel LDS tile
#define MIRROW 142       // 128 + 2*7 zero-pad ring
#define MIRSZ  (2*CCH*MIRROW*MIRROW)   // 1,008,200 bf16 elements

union BF16x16 { unsigned short u[16]; unsigned int w[8]; uint4 q[2]; v16bf v; };
union US8 { unsigned short u[8]; uint4 q; };

__device__ __forceinline__ unsigned short f2bf(float f) {
  __bf16 h = (__bf16)f;                       // native cvt on gfx1250
  union { __bf16 b; unsigned short u; } cv;
  cv.b = h;
  return cv.u;
}
__device__ __forceinline__ float softplusf(float x) {
  return fmaxf(x, 0.f) + __logf(1.f + __expf(-fabsf(x)));
}
__device__ __forceinline__ float sigmoidf(float x) {
  return __builtin_amdgcn_rcpf(1.f + __expf(-x));
}

// ---------------- 3D conv front-end (fp32 scalar; ~13% of FLOPs) --------------
__global__ void k_conv00(const float* __restrict__ x, const float* __restrict__ w,
                         float* __restrict__ out) {
  int i = blockIdx.x * 256 + threadIdx.x;          // 6,553,600 = 2*25*8*128*128
  int b  = i / 3276800;  int r  = i % 3276800;
  int co = r >> 17;      int r2 = r & 131071;
  int d  = r2 >> 14;     int y  = (r2 >> 7) & 127; int xx = r2 & 127;
  float acc = 0.f;
  for (int ci = 0; ci < 3; ++ci) {
    const float* xb = x + ((b * 3 + ci) * 8) * HW;
    const float* wb = w + (co * 3 + ci) * 343;
    for (int kd = 0; kd < 7; ++kd) {
      int zd = d + kd - 3; if ((unsigned)zd >= 8u) continue;
      for (int ky = 0; ky < 7; ++ky) {
        int zy = y + ky - 3; if ((unsigned)zy >= 128u) continue;
        const float* xr = xb + zd * HW + zy * Wd;
        const float* wr = wb + (kd * 7 + ky) * 7;
        for (int kx = 0; kx < 7; ++kx) {
          int zx = xx + kx - 3;
          if ((unsigned)zx < 128u) acc += xr[zx] * wr[kx];
        }
      }
    }
  }
  out[i] = acc;
}

__global__ void k_conv0(const float* __restrict__ a, const float* __restrict__ w,
                        float* __restrict__ out) {
  int i = blockIdx.x * 256 + threadIdx.x;          // 262,144 = 2*8*128*128
  int b  = i >> 17;  int r2 = i & 131071;
  int d  = r2 >> 14; int y  = (r2 >> 7) & 127; int xx = r2 & 127;
  float acc = 0.f;
  for (int ci = 0; ci < CCH; ++ci) {
    const float* ab = a + ((b * CCH + ci) * 8) * HW;
    const float* wb = w + ci * 343;
    for (int kd = 0; kd < 7; ++kd) {
      int zd = d + kd - 3; if ((unsigned)zd >= 8u) continue;
      for (int ky = 0; ky < 7; ++ky) {
        int zy = y + ky - 3; if ((unsigned)zy >= 128u) continue;
        const float* ar = ab + zd * HW + zy * Wd;
        const float* wr = wb + (kd * 7 + ky) * 7;
        for (int kx = 0; kx < 7; ++kx) {
          int zx = xx + kx - 3;
          if ((unsigned)zx < 128u) acc += ar[zx] * wr[kx];
        }
      }
    }
  }
  out[i] = acc;
}

__global__ void k_conv1sq(const float* __restrict__ a, const float* __restrict__ w,
                          const float* __restrict__ bias, float* __restrict__ out_t) {
  int i = blockIdx.x * 256 + threadIdx.x;          // 6,553,600
  int b  = i / 3276800;  int r  = i % 3276800;
  int co = r >> 17;      int r2 = r & 131071;
  int d  = r2 >> 14;     int y  = (r2 >> 7) & 127; int xx = r2 & 127;
  float acc = bias[co];
  const float* ab = a + (b * 8) * HW;
  const float* wb = w + co * 343;
  for (int kd = 0; kd < 7; ++kd) {
    int zd = d + kd - 3; if ((unsigned)zd >= 8u) continue;
    for (int ky = 0; ky < 7; ++ky) {
      int zy = y + ky - 3; if ((unsigned)zy >= 128u) continue;
      const float* ar = ab + zd * HW + zy * Wd;
      const float* wr = wb + (kd * 7 + ky) * 7;
      for (int kx = 0; kx < 7; ++kx) {
        int zx = xx + kx - 3;
        if ((unsigned)zx < 128u) acc += ar[zx] * wr[kx];
      }
    }
  }
  out_t[((d * 2 + b) * CCH + co) * HW + y * Wd + xx] = acc * acc;   // transpose + square
}

// ---- tap permutation: K slot -> tap index (pairs of horizontally adjacent taps)
__device__ __forceinline__ int tap_of_slot(int slot) {
  int q = slot >> 1, odd = slot & 1;
  if (q < 105) { int ky = q / 7, kx = 2 * (q % 7) + odd; return ky * 15 + kx; }
  if (q < 120 && odd == 0) { return (q - 105) * 15 + 14; }
  return -1;                                   // zero-weight slot
}

// ---------------- weight pre-pack into WMMA A-fragment order ------------------
// apack[mt][kg][ci][lane][j], 2*8*25*32*16 = 204,800 ushorts
__global__ void k_packw(const float* __restrict__ wgt, unsigned short* __restrict__ apack) {
  int i = blockIdx.x * 256 + threadIdx.x;
  if (i >= 204800) return;
  int j = i & 15;  int r = i >> 4;
  int lane = r & 31;  r >>= 5;
  int ci = r % 25;    r /= 25;
  int kg = r & 7;     int mt = r >> 3;
  int v = j >> 1, p = j & 1;
  int ka = (lane >> 4) * 16 + j;   // contiguous k = 16h + s (matches B)
  int slot = kg * 32 + ka;
  int tap = tap_of_slot(slot);
  int co = mt * 16 + (lane & 15);
  float w = (co < CCH && tap >= 0) ? wgt[(co * CCH + ci) * 225 + tap] : 0.f;
  apack[i] = f2bf(w);
}

// up[mt][lane][j], 2*32*16 = 1024 ushorts (1x1 conv weights)
__global__ void k_packu(const float* __restrict__ uw, unsigned short* __restrict__ up) {
  int i = blockIdx.x * 256 + threadIdx.x;
  if (i >= 1024) return;
  int j = i & 15;  int lane = (i >> 4) & 31;  int mt = i >> 9;
  int v = j >> 1, p = j & 1;
  int ka = (lane >> 4) * 16 + j;
  int co = mt * 16 + (lane & 15);
  float w = (co < CCH && ka < CCH) ? uw[co * CCH + ka] : 0.f;
  up[i] = f2bf(w);
}

// ---------------- 1x1 conv gate: WMMA GEMM 25x25 over pixels ------------------
__global__ void __launch_bounds__(128) k_gate(const float* __restrict__ src,
                                              const unsigned short* __restrict__ up,
                                              const float* __restrict__ ub,
                                              float* __restrict__ dstf,
                                              unsigned short* __restrict__ dmir,
                                              int mode) {
  __shared__ __align__(16) unsigned short tile[64 * 32];  // [px][ch pad 32]
  int blk = blockIdx.x;                 // 512 = 2 images * 256 pixel-tiles
  int b = blk >> 8;  int p0 = (blk & 255) * 64;
  int tid = threadIdx.x;
  for (int e = tid; e < 64 * 32; e += 128) {
    int ci = e >> 6, px = e & 63;
    float v = (ci < CCH) ? src[(b * CCH + ci) * HW + p0 + px] : 0.f;
    tile[px * 32 + ci] = f2bf(v);
  }
  __syncthreads();

  int wave = tid >> 5, lane = tid & 31;
  int col = lane & 15, kh = lane >> 4;
  int pix = p0 + wave * 16 + col;

  BF16x16 bfr, a0, a1;
  const unsigned int* trow = (const unsigned int*)&tile[(wave * 16 + col) * 32 + kh * 16];
#pragma unroll
  for (int v = 0; v < 8; ++v) bfr.w[v] = trow[v];
  const uint4* pa0 = (const uint4*)(up + lane * 16);
  const uint4* pa1 = (const uint4*)(up + 512 + lane * 16);
  a0.q[0] = pa0[0]; a0.q[1] = pa0[1];
  a1.q[0] = pa1[0]; a1.q[1] = pa1[1];

  v8f acc0 = {}; v8f acc1 = {};
  acc0 = __builtin_amdgcn_wmma_f32_16x16x32_bf16(false, a0.v, false, bfr.v, (short)0, acc0, false, false);
  acc1 = __builtin_amdgcn_wmma_f32_16x16x32_bf16(false, a1.v, false, bfr.v, (short)0, acc1, false, false);

  int py = pix >> 7, px = pix & 127;
#pragma unroll
  for (int v = 0; v < 8; ++v) {
    int row = kh * 8 + v;
    {
      int co = row;
      float g = sigmoidf(acc0[v] + ub[co]);
      if (mode == 0) {
        float o = src[(b * CCH + co) * HW + pix] * g;
        dmir[((b * CCH + co) * MIRROW + py + 7) * MIRROW + px + 7] = f2bf(o);
      } else {
        dstf[(b * CCH + co) * HW + pix] = g;
      }
    }
    int co = 16 + row;
    if (co < CCH) {
      float g = sigmoidf(acc1[v] + ub[co]);
      if (mode == 0) {
        float o = src[(b * CCH + co) * HW + pix] * g;
        dmir[((b * CCH + co) * MIRROW + py + 7) * MIRROW + px + 7] = f2bf(o);
      } else {
        dstf[(b * CCH + co) * HW + pix] = g;
      }
    }
  }
}

// ---------------- 15x15 conv, pad 7: WMMA implicit GEMM -----------------------
// LDS staging via Tensor Data Mover (contiguous 25x15x78 bf16 tile copy),
// fallback: dword copy loop. K reordered into tap pairs -> B = 8 dword LDS loads.
__global__ void __launch_bounds__(128) k_conv15(const unsigned short* __restrict__ mir,
                                                const unsigned short* __restrict__ apack,
                                                float* __restrict__ dst) {
  __shared__ __align__(16) unsigned short lds[CCH * CTILE + 8];  // 58,516 B
  __shared__ __align__(16) unsigned short toffp[128];
  int blk = blockIdx.x;                  // 512 = 2 images * 128 rows * 2 x-tiles
  int b = blk >> 8;  int r = blk & 255;
  int y0 = r >> 1;   int x0 = (r & 1) * 64;
  int tid = threadIdx.x;

  if (tid < 128) {
    int q = tid;
    unsigned short off;
    if (q < 105)      { int ky = q / 7;   off = (unsigned short)(ky * LDSW + 2 * (q % 7)); }
    else if (q < 120) { int ky = q - 105; off = (unsigned short)(ky * LDSW + 14); }
    else              off = 0;             // zero-weight slots: any finite data is fine
    toffp[q] = off;
  }
  if (tid < 8) lds[CCH * CTILE + tid] = 0; // tail pad for last single-tap overread

#if __has_builtin(__builtin_amdgcn_tensor_load_to_lds) && __has_builtin(__builtin_amdgcn_s_wait_tensorcnt)
  if (tid < 32) {  // wave 0 issues the TDM DMA: 25ch x 15rows x 78 bf16
    unsigned long long ga = (unsigned long long)mir
      + 2ull * (unsigned long long)(((b * CCH) * MIRROW + y0) * MIRROW + x0);
    unsigned ldsoff = (unsigned)(unsigned long long)&lds[0];   // flat low32 == LDS addr
    // D# group 0: count=1 | lds_addr | global_addr | type=2 (ISA 8.3)
    v4u g0 = { 1u, ldsoff, (unsigned)ga,
               (unsigned)((ga >> 32) & 0x1FFFFFFu) | (2u << 30) };
    // D# group 1 (ISA 8.4): data_size=1(2B); tensor_dim0/1 = 2^30 (no OOB);
    // tile 78 x 15 x 25; stride0=142, stride1=142*142=20164 (element units)
    v8i g1 = { (int)0x00010000u,                       // wg_mask=0, data_size=1
               (int)0x00000000u,                       // td0 low16<<16 (2^30 -> 0)
               (int)0x00004000u,                       // td0 hi16 | td1 low16<<16
               (int)((78u << 16) | 0x4000u),           // td1 hi16 | tile_dim0=78
               (int)((25u << 16) | 15u),               // tile_dim1=15 | tile_dim2=25
               (int)142u,                              // stride0[31:0]
               (int)((20164u & 0xFFFFu) << 16),        // stride0 hi=0 | stride1 lo16
               (int)(20164u >> 16) };                  // stride1[47:16]
    v4i g2 = { (int)0x40000000, 0, 0, 0 };             // tensor_dim2 big; tile_dim3=0
    v4i g3 = { 0, 0, 0, 0 };
    v8i g4 = { 0, 0, 0, 0, 0, 0, 0, 0 };               // extra group (6-arg toolchain)
    __builtin_amdgcn_tensor_load_to_lds(g0, g1, g2, g3, g4, 0);
    __builtin_amdgcn_s_wait_tensorcnt(0);
  }
#else
  for (int e = tid; e < 14625; e += 128) {   // 25 * 15 * 39 dwords
    int ci = e / 585;
    int r2 = e - ci * 585;
    int ky = r2 / 39;
    int lx = 2 * (r2 - ky * 39);
    const unsigned int* s = (const unsigned int*)
      &mir[((b * CCH + ci) * MIRROW + y0 + ky) * MIRROW + x0 + lx];
    *(unsigned int*)&lds[ci * CTILE + ky * LDSW + lx] = *s;
  }
#endif
  __syncthreads();

  int wave = tid >> 5, lane = tid & 31;
  int col = lane & 15, kh = lane >> 4;
  int xoff = wave * 16 + col;
  v8f acc0 = {}; v8f acc1 = {};

  for (int kg = 0; kg < 8; ++kg) {
    US8 tof;                                           // 8 pair offsets: 1x ds_load_b128
    tof.q = *(const uint4*)&toffp[kg * 16 + kh * 8];
    const unsigned short* apk0 = apack + (0 * 8 + kg) * 25 * 512 + lane * 16;
    const unsigned short* apk1 = apack + (1 * 8 + kg) * 25 * 512 + lane * 16;
    for (int ci = 0; ci < CCH; ++ci) {
      BF16x16 bfr, a0, a1;
      const unsigned short* ldsCi = &lds[ci * CTILE];
#pragma unroll
      for (int v = 0; v < 8; ++v) {                    // 8 dword pair loads
        unsigned int d;
        __builtin_memcpy(&d, ldsCi + tof.u[v] + xoff, 4);
        bfr.w[v] = d;
      }
      const uint4* pa0 = (const uint4*)(apk0 + ci * 512);
      const uint4* pa1 = (const uint4*)(apk1 + ci * 512);
      a0.q[0] = pa0[0]; a0.q[1] = pa0[1];
      a1.q[0] = pa1[0]; a1.q[1] = pa1[1];
      acc0 = __builtin_amdgcn_wmma_f32_16x16x32_bf16(false, a0.v, false, bfr.v, (short)0, acc0, false, false);
      acc1 = __builtin_amdgcn_wmma_f32_16x16x32_bf16(false, a1.v, false, bfr.v, (short)0, acc1, false, false);
    }
  }

  int x = x0 + wave * 16 + col;
#pragma unroll
  for (int v = 0; v < 8; ++v) {
    int row = kh * 8 + v;
    dst[(b * CCH + row) * HW + y0 * Wd + x] = acc0[v];
    if (16 + row < CCH)
      dst[(b * CCH + 16 + row) * HW + y0 * Wd + x] = acc1[v];
  }
}

// ---------------- BatchNorm batch stats (biased var, eps=1e-3) ----------------
__global__ void k_bnstats(const float* __restrict__ src, float* __restrict__ stat) {
  __shared__ float s1[256], s2[256];
  int c = blockIdx.x, tid = threadIdx.x;
  float a = 0.f, q = 0.f;
  for (int i = tid; i < 2 * HW; i += 256) {
    int b = i >> 14, p = i & (HW - 1);
    float v = src[(b * CCH + c) * HW + p];
    a += v; q += v * v;
  }
  s1[tid] = a; s2[tid] = q; __syncthreads();
  for (int s = 128; s > 0; s >>= 1) {
    if (tid < s) { s1[tid] += s1[tid + s]; s2[tid] += s2[tid + s]; }
    __syncthreads();
  }
  if (tid == 0) {
    float m = s1[0] / (2.f * HW);
    float var = s2[0] / (2.f * HW) - m * m;
    stat[2 * c] = m;
    stat[2 * c + 1] = rsqrtf(var + 1e-3f);
  }
}

// ---------------- elementwise recurrence steps --------------------------------
__global__ void k_step1(const float* __restrict__ y, const float* __restrict__ craw,
                        const float* __restrict__ stat, const float* __restrict__ h,
                        const float* __restrict__ bn1w, const float* __restrict__ bn1b,
                        const float* __restrict__ alpha, const float* __restrict__ mu,
                        float* __restrict__ ns1, unsigned short* __restrict__ nmir) {
  int i = blockIdx.x * 256 + threadIdx.x;           // 819,200
  int bc = i >> 14;
  int c = bc % CCH;
  float c1 = (craw[i] - stat[2 * c]) * stat[2 * c + 1] * bn1w[c] + bn1b[c];
  float hv = h[i];
  float n = softplusf(y[i] - softplusf(c1 * (alpha[c] * hv + mu[c])));
  ns1[i] = n;
  int p = i & (HW - 1);
  nmir[(bc * MIRROW + (p >> 7) + 7) * MIRROW + (p & 127) + 7] = f2bf(n);
}

__global__ void k_step2(const float* __restrict__ ns1, const float* __restrict__ craw,
                        const float* __restrict__ stat, const float* __restrict__ g2,
                        const float* __restrict__ bn3w, const float* __restrict__ bn3b,
                        const float* __restrict__ kap, const float* __restrict__ gam,
                        const float* __restrict__ wmx, float* __restrict__ h) {
  int i = blockIdx.x * 256 + threadIdx.x;
  int c = (i >> 14) % CCH;
  float c2 = (craw[i] - stat[2 * c]) * stat[2 * c + 1] * bn3w[c] + bn3b[c];
  float n = ns1[i], g = g2[i];
  float h2 = softplusf(kap[c] * n + gam[c] * c2 + wmx[c] * n * c2);
  h[i] = softplusf((1.f - g) * h[i] + g * h2);
}

__global__ void k_zero(float* p, int n) {
  int i = blockIdx.x * 256 + threadIdx.x;
  if (i < n) p[i] = 0.f;
}
__global__ void k_zero16(unsigned short* p, int n) {
  int i = blockIdx.x * 256 + threadIdx.x;
  if (i < n) p[i] = 0;
}

// ---------------- BN-out + avgpool2 + FC + sigmoid ----------------------------
__global__ void k_fc(const float* __restrict__ h, const float* __restrict__ stat,
                     const float* __restrict__ bw, const float* __restrict__ bb,
                     const float* __restrict__ fw, const float* __restrict__ fb,
                     float* __restrict__ out) {
  __shared__ float s[256];
  int row = blockIdx.x, tid = threadIdx.x;
  float acc = 0.f;
  for (int j = tid; j < 204800; j += 256) {
    int b = j / 102400; int r = j % 102400;
    int c = r >> 12;    int rr = r & 4095;
    int yy = rr >> 6,   xx = rr & 63;
    const float* base = h + (b * CCH + c) * HW + (yy * 2) * Wd + xx * 2;
    float pooled = 0.25f * (base[0] + base[1] + base[Wd] + base[Wd + 1]);
    float v = (pooled - stat[2 * c]) * stat[2 * c + 1] * bw[c] + bb[c];
    acc += v * fw[row * 204800 + j];
  }
  s[tid] = acc; __syncthreads();
  for (int st = 128; st > 0; st >>= 1) {
    if (tid < st) s[tid] += s[tid + st];
    __syncthreads();
  }
  if (tid == 0) out[row] = sigmoidf(s[0] + fb[row]);
}

// ---------------- host orchestration ------------------------------------------
extern "C" void kernel_launch(void* const* d_in, const int* in_sizes, int n_in,
                              void* d_out, int out_size, void* d_ws, size_t ws_size,
                              hipStream_t stream) {
  const float* x       = (const float*)d_in[0];
  const float* conv00w = (const float*)d_in[1];
  const float* conv0w  = (const float*)d_in[2];
  const float* conv1w  = (const float*)d_in[3];
  const float* conv1b  = (const float*)d_in[4];
  const float* u1w     = (const float*)d_in[5];
  const float* u1b     = (const float*)d_in[6];
  const float* u2w     = (const float*)d_in[7];
  const float* u2b     = (const float*)d_in[8];
  const float* winh    = (const float*)d_in[9];
  const float* wexc    = (const float*)d_in[10];
  const float* alpha   = (const float*)d_in[11];
  const float* mu      = (const float*)d_in[12];
  const float* gammap  = (const float*)d_in[13];
  const float* kappa   = (const float*)d_in[14];
  const float* wmix    = (const float*)d_in[15];
  const float* bn1w    = (const float*)d_in[16];
  const float* bn1b    = (const float*)d_in[17];
  const float* bn3w    = (const float*)d_in[18];
  const float* bn3b    = (const float*)d_in[19];
  const float* bnow    = (const float*)d_in[20];
  const float* bnob    = (const float*)d_in[21];
  const float* fc4w    = (const float*)d_in[22];
  const float* fc4b    = (const float*)d_in[23];
  float* out = (float*)d_out;
  float* ws  = (float*)d_ws;

  // float-offset workspace layout (total ~53.5 MB)
  float* out_t = ws;                          // [8][2][25][128][128]
  float* tmpA  = ws + 6553600;                // conv00 out; reused after conv0
  float* tmpB  = ws + 13107200;               // conv0 out [2][8][128][128]
  float* h     = tmpA;                        // [2][25][128][128]
  float* ns1   = tmpA + 1638400;
  float* g2    = tmpA + 2457600;
  float* craw  = tmpA + 3276800;
  float* stat  = tmpA + 4096000;              // 50 floats
  unsigned short* pwinh = (unsigned short*)(tmpA + 4100000);  // 204,800 us
  unsigned short* pwexc = pwinh + 204800;
  unsigned short* pu1   = pwexc + 204800;
  unsigned short* pu2   = pu1 + 1024;
  unsigned short* mirHG  = (unsigned short*)(tmpA + 4310000); // 1,008,200 us
  unsigned short* mirNS1 = (unsigned short*)(tmpA + 4815000); // 1,008,200 us

  // front-end
  k_conv00 <<<25600, 256, 0, stream>>>(x, conv00w, tmpA);
  k_conv0  <<<1024,  256, 0, stream>>>(tmpA, conv0w, tmpB);
  // tmpA region is free after k_conv0 (stream order)
  k_packw  <<<800, 256, 0, stream>>>(winh, pwinh);
  k_packw  <<<800, 256, 0, stream>>>(wexc, pwexc);
  k_packu  <<<4,   256, 0, stream>>>(u1w, pu1);
  k_packu  <<<4,   256, 0, stream>>>(u2w, pu2);
  k_zero16 <<<3939, 256, 0, stream>>>(mirHG, MIRSZ);
  k_zero16 <<<3939, 256, 0, stream>>>(mirNS1, MIRSZ);
  k_conv1sq<<<25600, 256, 0, stream>>>(tmpB, conv1w, conv1b, out_t);

  k_zero<<<3200, 256, 0, stream>>>(h, 819200);

  // hGRU recurrence, 8 timesteps
  for (int t = 0; t < 8; ++t) {
    const float* y = out_t + t * 819200;
    k_gate   <<<512, 128, 0, stream>>>(h, pu1, u1b, nullptr, mirHG, 0);
    k_conv15 <<<512, 128, 0, stream>>>(mirHG, pwinh, craw);
    k_bnstats<<<25,  256, 0, stream>>>(craw, stat);
    k_step1  <<<3200,256, 0, stream>>>(y, craw, stat, h, bn1w, bn1b, alpha, mu, ns1, mirNS1);
    k_gate   <<<512, 128, 0, stream>>>(ns1, pu2, u2b, g2, nullptr, 1);
    k_conv15 <<<512, 128, 0, stream>>>(mirNS1, pwexc, craw);
    k_bnstats<<<25,  256, 0, stream>>>(craw, stat);
    k_step2  <<<3200,256, 0, stream>>>(ns1, craw, stat, g2, bn3w, bn3b, kappa, gammap, wmix, h);
  }

  // readout
  k_bnstats<<<25, 256, 0, stream>>>(h, stat);
  k_fc     <<<2,  256, 0, stream>>>(h, stat, bnow, bnob, fc4w, fc4b, out);
}
